// DMPNN_11647951307194
// MI455X (gfx1250) — compile-verified, matrix operand order checked by reading, not run on hardware
//
#include <hip/hip_runtime.h>
#include <math.h>

// Problem constants (match reference)
#define NN  50000
#define EE  400000
#define ELG 1600000
#define BB  2000
#define FF  64
#define EDD 16
#define TT  6
#define SND 384

typedef __attribute__((ext_vector_type(16))) _Float16 v16h;
typedef __attribute__((ext_vector_type(8)))  float    v8f;

enum { ACT_NONE = 0, ACT_RELU = 1, ACT_TANH = 2 };

// ---------------------------------------------------------------------------
// WMMA GEMM:  out[M,NO] = act(alpha * (A[M,K] @ W[NO,K]^T [+ bias] [+ res]))
// A, W row-major fp32; fragments converted to f16 on the fly.
// K / bias / res / act are compile-time so the K-loop and epilogue carry no
// runtime branches. A rows are clamped (not predicated) so fragment loads are
// unconditional and EXEC stays all-1 for WMMA. Each wave computes a
// 16 x (16*NT) strip: one A-fragment feeds NT back-to-back WMMAs.
// ---------------------------------------------------------------------------
template <int K>
__device__ __forceinline__ v16h frag_ld(const float* __restrict__ base,
                                        long long row, int kb, int hi) {
  v16h f;
#pragma unroll
  for (int c = 0; c < 4; ++c) {
    const int cbase = ((c & 2) ? 16 : 0);        // K-offset of this chunk pair
    const int kk = kb + cbase + hi * 8 + (c & 1) * 4;
    float4 v = make_float4(0.f, 0.f, 0.f, 0.f);
    // (K%32==0) folds true for K=64/384; for K=16 (kb==0, single trip) the
    // per-chunk bound folds per c. No divergent branches are emitted.
    if ((K % 32 == 0) || (kb + cbase + 16 <= K))
      v = *(const float4*)(base + row * (long long)K + kk);
    f[c * 4 + 0] = (_Float16)v.x;
    f[c * 4 + 1] = (_Float16)v.y;
    f[c * 4 + 2] = (_Float16)v.z;
    f[c * 4 + 3] = (_Float16)v.w;
  }
  return f;
}

template <int K, int NT, bool HB, bool HR, int ACT>
__global__ __launch_bounds__(128) void wmma_gemm_kernel(
    const float* __restrict__ A, const float* __restrict__ W,
    const float* __restrict__ bias, const float* __restrict__ res,
    float* __restrict__ out, int M, int NO, float alpha) {
  const int lane   = threadIdx.x & 31;
  const int wave   = threadIdx.x >> 5;
  const int lane15 = lane & 15;
  const int hi     = lane >> 4;
  const int n0     = blockIdx.x * 16 * NT;
  const long long m0 = ((long long)blockIdx.y * 4 + wave) * 16;
  if (m0 >= M) return;  // wave-uniform: EXEC all-1 for WMMA below

  // Clamp A row: OOB rows only feed accumulator rows masked at store.
  long long arow = m0 + lane15;
  if (arow >= M) arow = M - 1;

  v8f acc[NT];
#pragma unroll
  for (int nt = 0; nt < NT; ++nt) acc[nt] = (v8f){};

#pragma unroll 2
  for (int kb = 0; kb < K; kb += 32) {
    v16h af = frag_ld<K>(A, arow, kb, hi);
    v16h bf[NT];
#pragma unroll
    for (int nt = 0; nt < NT; ++nt)
      bf[nt] = frag_ld<K>(W, (long long)(n0 + nt * 16 + lane15), kb, hi);
#pragma unroll
    for (int nt = 0; nt < NT; ++nt)
      acc[nt] = __builtin_amdgcn_wmma_f32_16x16x32_f16(
          /*neg_a=*/false, af, /*neg_b=*/false, bf[nt],
          /*c_mod=*/(short)0, acc[nt], /*reuse_a=*/false, /*reuse_b=*/false);
  }

  const bool fullM = (m0 + 16 <= (long long)M);  // wave-uniform fast path
#pragma unroll
  for (int nt = 0; nt < NT; ++nt) {
    const int n = n0 + nt * 16 + lane15;
    const float bv = HB ? bias[n] : 0.f;
    if (fullM) {
#pragma unroll
      for (int r = 0; r < 8; ++r) {
        const long long m = m0 + r + hi * 8;
        float v = acc[nt][r] + bv;
        if (HR) v += res[m * (long long)NO + n];
        v *= alpha;
        if (ACT == ACT_RELU)      v = fmaxf(v, 0.f);
        else if (ACT == ACT_TANH) v = tanhf(v);
        out[m * (long long)NO + n] = v;
      }
    } else {
#pragma unroll
      for (int r = 0; r < 8; ++r) {
        const long long m = m0 + r + hi * 8;
        if (m >= (long long)M) continue;
        float v = acc[nt][r] + bv;
        if (HR) v += res[m * (long long)NO + n];
        v *= alpha;
        if (ACT == ACT_RELU)      v = fmaxf(v, 0.f);
        else if (ACT == ACT_TANH) v = tanhf(v);
        out[m * (long long)NO + n] = v;
      }
    }
  }
}

// ---------------------------------------------------------------------------
// Elementwise / scatter kernels
// ---------------------------------------------------------------------------
__global__ void fill_kernel(float* __restrict__ p, long long n, float val) {
  long long i = (long long)blockIdx.x * blockDim.x + threadIdx.x;
  long long stride = (long long)gridDim.x * blockDim.x;
  for (; i < n; i += stride) p[i] = val;
}

// ef[e,:] = (ef[e,:] + xu[src[e],:] + xv[dst[e],:]) / 3
__global__ void edge_feat_kernel(float* __restrict__ ef,
                                 const float* __restrict__ xu,
                                 const float* __restrict__ xv,
                                 const int* __restrict__ src,
                                 const int* __restrict__ dst) {
  long long n = (long long)EE * 16;
  long long i = (long long)blockIdx.x * blockDim.x + threadIdx.x;
  long long stride = (long long)gridDim.x * blockDim.x;
  for (; i < n; i += stride) {
    long long e  = i >> 4;
    int       f4 = (int)(i & 15) * 4;
    int s = src[e], d = dst[e];
    float4 ea = *(float4*)(ef + e * 64 + f4);
    float4 u  = *(const float4*)(xu + (long long)s * 64 + f4);
    float4 v  = *(const float4*)(xv + (long long)d * 64 + f4);
    float4 r;
    r.x = (ea.x + u.x + v.x) * (1.f / 3.f);
    r.y = (ea.y + u.y + v.y) * (1.f / 3.f);
    r.z = (ea.z + u.z + v.z) * (1.f / 3.f);
    r.w = (ea.w + u.w + v.w) * (1.f / 3.f);
    *(float4*)(ef + e * 64 + f4) = r;
  }
}

// agg[ldst[le],:] += srcbuf[lsrc[le],:]
__global__ void scatter_add_kernel(const float* __restrict__ srcbuf,
                                   float* __restrict__ agg,
                                   const int* __restrict__ lsrc,
                                   const int* __restrict__ ldst) {
  long long n = (long long)ELG * 16;
  long long i = (long long)blockIdx.x * blockDim.x + threadIdx.x;
  long long stride = (long long)gridDim.x * blockDim.x;
  for (; i < n; i += stride) {
    long long le = i >> 4;
    int       f4 = (int)(i & 15) * 4;
    int s = lsrc[le], d = ldst[le];
    float4 v = *(const float4*)(srcbuf + (long long)s * 64 + f4);
    float* dp = agg + (long long)d * 64 + f4;
    atomicAdd(dp + 0, v.x);
    atomicAdd(dp + 1, v.y);
    atomicAdd(dp + 2, v.z);
    atomicAdd(dp + 3, v.w);
    if (i + stride < n) {  // gfx1250 global_prefetch_b8 for next gather row
      long long le2 = (i + stride) >> 4;
      __builtin_prefetch(srcbuf + (long long)lsrc[le2] * 64, 0, 0);
    }
  }
}

__global__ void add_kernel(const float* __restrict__ a, const float* __restrict__ b,
                           float* __restrict__ o, long long n) {
  long long i = (long long)blockIdx.x * blockDim.x + threadIdx.x;
  long long stride = (long long)gridDim.x * blockDim.x;
  for (; i < n; i += stride) o[i] = a[i] + b[i];
}

// x_conv[e] = dot(agg[e], w_rel) + b_rel + dot(out[e], w_root)
__global__ void xconv_kernel(const float* __restrict__ outb,
                             const float* __restrict__ agg,
                             const float* __restrict__ w_rel,
                             const float* __restrict__ b_rel,
                             const float* __restrict__ w_root,
                             float* __restrict__ xcv) {
  long long e = (long long)blockIdx.x * blockDim.x + threadIdx.x;
  if (e >= EE) return;
  float s = b_rel[0];
#pragma unroll 8
  for (int f = 0; f < 64; ++f)
    s += agg[e * 64 + f] * w_rel[f] + outb[e * 64 + f] * w_root[f];
  xcv[e] = s;
}

__device__ __forceinline__ void atomicMaxF(float* addr, float v) {
  if (v >= 0.f) atomicMax((int*)addr, __float_as_int(v));
  else          atomicMin((unsigned int*)addr, (unsigned int)__float_as_int(v));
}

__global__ void segmax_kernel(const float* __restrict__ xcv,
                              const int* __restrict__ batch,
                              float* __restrict__ m) {
  long long e = (long long)blockIdx.x * blockDim.x + threadIdx.x;
  if (e >= EE) return;
  atomicMaxF(&m[batch[e]], xcv[e]);
}

__global__ void segexp_kernel(const float* __restrict__ xcv,
                              const int* __restrict__ batch,
                              const float* __restrict__ m,
                              float* __restrict__ ev, float* __restrict__ s) {
  long long e = (long long)blockIdx.x * blockDim.x + threadIdx.x;
  if (e >= EE) return;
  int b = batch[e];
  float t = expf(xcv[e] - m[b]);
  ev[e] = t;
  atomicAdd(&s[b], t);
}

// g[b,:] += out[e,:] * (ev[e] / (s[b] + eps))
__global__ void gpool_kernel(const float* __restrict__ outb,
                             const float* __restrict__ ev,
                             const float* __restrict__ s,
                             const int* __restrict__ batch,
                             float* __restrict__ g) {
  long long n = (long long)EE * 16;
  long long i = (long long)blockIdx.x * blockDim.x + threadIdx.x;
  long long stride = (long long)gridDim.x * blockDim.x;
  for (; i < n; i += stride) {
    long long e  = i >> 4;
    int       f4 = (int)(i & 15) * 4;
    int b = batch[e];
    float sc = ev[e] / (s[b] + 1e-16f);
    float4 v = *(const float4*)(outb + e * 64 + f4);
    float* gp = g + (long long)b * 64 + f4;
    atomicAdd(gp + 0, v.x * sc);
    atomicAdd(gp + 1, v.y * sc);
    atomicAdd(gp + 2, v.z * sc);
    atomicAdd(gp + 3, v.w * sc);
  }
}

__global__ void cvt16_kernel(const float* __restrict__ src, _Float16* __restrict__ dst,
                             long long n) {
  long long i = (long long)blockIdx.x * blockDim.x + threadIdx.x;
  long long stride = (long long)gridDim.x * blockDim.x;
  for (; i < n; i += stride) dst[i] = (_Float16)src[i];
}

// per-batch attention over T iterations
__global__ void scores_kernel(const float* __restrict__ gouts,   // [T,B,64]
                              const float* __restrict__ a,       // [64,T]
                              const float* __restrict__ a_bias,  // [T]
                              float* __restrict__ scores) {      // [B,T]
  int b = blockIdx.x * blockDim.x + threadIdx.x;
  if (b >= BB) return;
  float sc[TT];
#pragma unroll
  for (int t = 0; t < TT; ++t) {
    float s = a_bias[t];
    for (int f = 0; f < 64; ++f)
      s += gouts[(long long)t * BB * 64 + (long long)b * 64 + f] * a[f * TT + t];
    sc[t] = s;
  }
  float mx = sc[0];
#pragma unroll
  for (int t = 1; t < TT; ++t) mx = fmaxf(mx, sc[t]);
  float sum = 0.f;
#pragma unroll
  for (int t = 0; t < TT; ++t) { sc[t] = expf(sc[t] - mx); sum += sc[t]; }
#pragma unroll
  for (int t = 0; t < TT; ++t) scores[b * TT + t] = sc[t] / sum;
}

// hsum[dst[e],f] += sum_t outs16[t][e,f] * scores[batch[e],t]
__global__ void outw_scatter_kernel(const _Float16* __restrict__ outs16,
                                    const float* __restrict__ scores,
                                    const int* __restrict__ batch,
                                    const int* __restrict__ dst,
                                    float* __restrict__ hsum) {
  long long n = (long long)EE * 64;
  long long i = (long long)blockIdx.x * blockDim.x + threadIdx.x;
  long long stride = (long long)gridDim.x * blockDim.x;
  for (; i < n; i += stride) {
    long long e = i >> 6;
    int b = batch[e];
    float w = 0.f;
#pragma unroll
    for (int t = 0; t < TT; ++t)
      w += (float)outs16[(long long)t * EE * 64 + i] * scores[b * TT + t];
    atomicAdd(&hsum[(long long)dst[e] * 64 + (int)(i & 63)], w);
  }
}

__global__ void addx_kernel(const float* __restrict__ x, float* __restrict__ h,
                            long long n) {
  long long i = (long long)blockIdx.x * blockDim.x + threadIdx.x;
  long long stride = (long long)gridDim.x * blockDim.x;
  for (; i < n; i += stride) h[i] += x[i];
}

// ---------------------------------------------------------------------------
// Host orchestration
// ---------------------------------------------------------------------------
template <int K, bool HB, bool HR, int ACT>
static inline void gemm_t(const float* A, const float* W, const float* bias,
                          const float* res, float* out, int M, int NO,
                          float alpha, hipStream_t st) {
  constexpr int NT = 4;            // 16*NT = 64-wide N strip; NO % 64 == 0 here
  dim3 grid(NO / (16 * NT), (unsigned)((M + 63) / 64));
  wmma_gemm_kernel<K, NT, HB, HR, ACT><<<grid, 128, 0, st>>>(A, W, bias, res, out,
                                                             M, NO, alpha);
}

extern "C" void kernel_launch(void* const* d_in, const int* in_sizes, int n_in,
                              void* d_out, int out_size, void* d_ws, size_t ws_size,
                              hipStream_t stream) {
  const float* x          = (const float*)d_in[0];
  const float* edge_attr  = (const float*)d_in[1];
  const int*   edge_index = (const int*)d_in[2];    // [2,E]
  const int*   lg_index   = (const int*)d_in[3];    // [2,ELG]
  const int*   batch      = (const int*)d_in[4];    // [E]
  const float* W_u        = (const float*)d_in[5];
  const float* W_v        = (const float*)d_in[6];
  const float* W_e        = (const float*)d_in[7];
  const float* cw_rel     = (const float*)d_in[8];
  const float* cb_rel     = (const float*)d_in[9];
  const float* cw_root    = (const float*)d_in[10];
  const float* a_att      = (const float*)d_in[11];
  const float* W_gout     = (const float*)d_in[12];
  const float* b_gout     = (const float*)d_in[13];
  const float* a_bias     = (const float*)d_in[14];
  const float* lb1_w = (const float*)d_in[15]; const float* lb1_b = (const float*)d_in[16];
  const float* lb2_w = (const float*)d_in[17]; const float* lb2_b = (const float*)d_in[18];
  const float* lb3_w = (const float*)d_in[19]; const float* lb3_b = (const float*)d_in[20];
  const float* lb4_w = (const float*)d_in[21]; const float* lb4_b = (const float*)d_in[22];
  const float* lb5_w = (const float*)d_in[23]; const float* lb5_b = (const float*)d_in[24];

  const int* src  = edge_index;
  const int* dst  = edge_index + EE;
  const int* lsrc = lg_index;
  const int* ldst = lg_index + ELG;

  // workspace bump allocator
  char* wp = (char*)d_ws;
  auto alloc = [&](size_t bytes) -> char* {
    char* r = wp;
    wp += (bytes + 255) & ~(size_t)255;
    return r;
  };
  float*    xu     = (float*)alloc((size_t)NN * 64 * 4);
  float*    xv     = (float*)alloc((size_t)NN * 64 * 4);
  float*    ef     = (float*)alloc((size_t)EE * 64 * 4);
  float*    outb   = (float*)alloc((size_t)EE * 64 * 4);
  float*    agg    = (float*)alloc((size_t)EE * 64 * 4);
  float*    xcv    = (float*)alloc((size_t)EE * 4);
  float*    ev     = (float*)alloc((size_t)EE * 4);
  float*    mbuf   = (float*)alloc((size_t)BB * 4);
  float*    sbuf   = (float*)alloc((size_t)BB * 4);
  float*    gbuf   = (float*)alloc((size_t)BB * 64 * 4);
  float*    gouts  = (float*)alloc((size_t)TT * BB * 64 * 4);
  float*    scores = (float*)alloc((size_t)BB * TT * 4);
  float*    hbuf   = (float*)alloc((size_t)NN * 64 * 4);
  float*    h1     = (float*)alloc((size_t)NN * SND * 4);
  float*    t2     = (float*)alloc((size_t)NN * SND * 4);
  float*    h2     = (float*)alloc((size_t)NN * SND * 4);
  _Float16* outs16 = (_Float16*)alloc((size_t)TT * EE * 64 * 2);
  if ((size_t)(wp - (char*)d_ws) > ws_size) return;  // insufficient workspace

  const dim3 tb(256);
  const dim3 gsb(4096);  // grid-stride blocks for big elementwise/scatter work
  const dim3 eb((EE + 255) / 256);

  // initial edge features: ef = (edge_attr@W_e^T + xu[src] + xv[dst]) / 3
  gemm_t<FF, false, false, ACT_NONE>(x, W_u, nullptr, nullptr, xu, NN, FF, 1.f, stream);
  gemm_t<FF, false, false, ACT_NONE>(x, W_v, nullptr, nullptr, xv, NN, FF, 1.f, stream);
  gemm_t<EDD, false, false, ACT_NONE>(edge_attr, W_e, nullptr, nullptr, ef, EE, FF, 1.f, stream);
  edge_feat_kernel<<<gsb, tb, 0, stream>>>(ef, xu, xv, src, dst);

  for (int t = 0; t < TT; ++t) {
    const float* msg_src = (t == 0) ? ef : outb;
    // out = ef + segsum(out[lsrc] -> ldst)
    fill_kernel<<<gsb, tb, 0, stream>>>(agg, (long long)EE * 64, 0.f);
    scatter_add_kernel<<<gsb, tb, 0, stream>>>(msg_src, agg, lsrc, ldst);
    add_kernel<<<gsb, tb, 0, stream>>>(ef, agg, outb, (long long)EE * 64);
    // conv_agg = segsum(out[lsrc] -> ldst)
    fill_kernel<<<gsb, tb, 0, stream>>>(agg, (long long)EE * 64, 0.f);
    scatter_add_kernel<<<gsb, tb, 0, stream>>>(outb, agg, lsrc, ldst);
    // x_conv, segment softmax over batch, attention pool
    xconv_kernel<<<eb, tb, 0, stream>>>(outb, agg, cw_rel, cb_rel, cw_root, xcv);
    fill_kernel<<<8, tb, 0, stream>>>(mbuf, BB, -INFINITY);
    fill_kernel<<<8, tb, 0, stream>>>(sbuf, BB, 0.f);
    segmax_kernel<<<eb, tb, 0, stream>>>(xcv, batch, mbuf);
    segexp_kernel<<<eb, tb, 0, stream>>>(xcv, batch, mbuf, ev, sbuf);
    fill_kernel<<<512, tb, 0, stream>>>(gbuf, (long long)BB * 64, 0.f);
    gpool_kernel<<<gsb, tb, 0, stream>>>(outb, ev, sbuf, batch, gbuf);
    // gout_t = tanh(g @ W_gout^T + b_gout)
    gemm_t<FF, true, false, ACT_TANH>(gbuf, W_gout, b_gout, nullptr,
                                      gouts + (size_t)t * BB * 64, BB, FF, 1.f, stream);
    // stash out_t in f16
    cvt16_kernel<<<gsb, tb, 0, stream>>>(outb, outs16 + (size_t)t * EE * 64,
                                         (long long)EE * 64);
  }

  // attention over T, weighted edge sum scattered to nodes, residual with x
  scores_kernel<<<(BB + 255) / 256, tb, 0, stream>>>(gouts, a_att, a_bias, scores);
  fill_kernel<<<gsb, tb, 0, stream>>>(hbuf, (long long)NN * 64, 0.f);
  outw_scatter_kernel<<<gsb, tb, 0, stream>>>(outs16, scores, batch, dst, hbuf);
  addx_kernel<<<gsb, tb, 0, stream>>>(x, hbuf, (long long)NN * 64);

  // LinearBlock
  gemm_t<FF,  true, false, ACT_RELU>(hbuf, lb1_w, lb1_b, nullptr, h1, NN, SND, 1.f, stream);
  gemm_t<SND, true, false, ACT_RELU>(h1, lb2_w, lb2_b, nullptr, t2, NN, SND, 1.f, stream);
  gemm_t<SND, true, true,  ACT_NONE>(t2, lb3_w, lb3_b, h1, h2, NN, SND, 0.5f, stream);
  gemm_t<SND, true, true,  ACT_NONE>(h2, lb4_w, lb4_b, h2, t2, NN, SND, 0.5f, stream);
  gemm_t<SND, true, false, ACT_RELU>(t2, lb5_w, lb5_b, nullptr, (float*)d_out, NN, FF, 1.f,
                                     stream);
}
